// HyperGCN_62242666053890
// MI455X (gfx1250) — compile-verified, matrix operand order checked by reading, not run on hardware
//
#include <hip/hip_runtime.h>
#include <hip/hip_bf16.h>

// ---------------- constants ----------------
#define NV    200000
#define NE    400000
#define INCH  500
#define HID   128
#define NCLS  40
#define HSIZE (1u << 21)
#define HMASK (HSIZE - 1u)

// ---------------- WMMA types ----------------
typedef __attribute__((ext_vector_type(16))) __bf16 v16bf;
typedef __attribute__((ext_vector_type(8)))  __bf16 bf16x8;
typedef __attribute__((ext_vector_type(8)))  float  v8f;

union Frag16 { v16bf v; bf16x8 h[2]; };

__device__ __forceinline__ unsigned short f32_to_bf16_bits(float f) {
  unsigned int u = __float_as_uint(f);
  u += 0x7FFFu + ((u >> 16) & 1u);   // round-to-nearest-even
  return (unsigned short)(u >> 16);
}

// ---------------- weight convert: f32 [Korig,Nin] -> bf16 transposed [Npad,Kpad] ----------------
__global__ void cvt_wT_kernel(const float* __restrict__ W, unsigned short* __restrict__ WbT,
                              int Korig, int Kpad, int Nin, int Npad) {
  int idx = blockIdx.x * blockDim.x + threadIdx.x;
  if (idx >= Npad * Kpad) return;
  int n = idx / Kpad, k = idx % Kpad;
  float v = (k < Korig && n < Nin) ? W[(size_t)k * Nin + n] : 0.0f;
  WbT[idx] = f32_to_bf16_bits(v);
}

// ---------------- tiled bf16 WMMA GEMM ----------------
// C[M,Nout] = A[M,Korig] * B[Korig,Nout], with BT = bf16 B transposed [NT*16][Kpad].
// block = 256 threads (8 waves). Block tile: 256 rows x NT*16 cols.
// Wave w owns rows [w*32, w*32+32): two 16-row M-tiles sharing each B fragment.
// A tile staged in LDS (f32->bf16), double buffered; B fragments read from global (L2-resident).
template <int NT>
__global__ __launch_bounds__(256)
void gemm_bf16_wmma(const float* __restrict__ A, const unsigned short* __restrict__ BT,
                    float* __restrict__ C, int M, int Korig, int Kpad, int Nout) {
  __shared__ __align__(16) unsigned short As[2][256][32];

  const int tid  = threadIdx.x;
  const int lane = tid & 31;
  const int wave = tid >> 5;            // 0..7
  const int rowBase = blockIdx.x * 256;

  v8f acc0[NT], acc1[NT];
  for (int t = 0; t < NT; ++t)
    for (int i = 0; i < 8; ++i) { acc0[t][i] = 0.0f; acc1[t][i] = 0.0f; }

  const int half = lane >> 4;           // 0 or 1
  const int l15  = lane & 15;
  const int ka0  = half * 8;            // K sub-chunk base per ISA bf16 A/B layout
  const int ka1  = ka0 + 16;

  // stage one 256x32 A tile (f32 -> bf16) into LDS buffer `buf`
  auto loadA = [&](int k0, int buf) {
    #pragma unroll
    for (int i = 0; i < 8; ++i) {
      int s  = tid + i * 256;           // float4 slot 0..2047
      int r  = s >> 3;                  // 8 slots per 32-wide row
      int kc = (s & 7) * 4;
      int gr = rowBase + r;
      int gk = k0 + kc;
      float vx = 0.0f, vy = 0.0f, vz = 0.0f, vw = 0.0f;
      if (gr < M) {
        const float* ap = A + (size_t)gr * Korig;
        if (gk + 3 < Korig) {
          float4 t4 = *(const float4*)(ap + gk);
          vx = t4.x; vy = t4.y; vz = t4.z; vw = t4.w;
        } else {
          if (gk     < Korig) vx = ap[gk];
          if (gk + 1 < Korig) vy = ap[gk + 1];
          if (gk + 2 < Korig) vz = ap[gk + 2];
          if (gk + 3 < Korig) vw = ap[gk + 3];
        }
      }
      uint2 pk;
      pk.x = (unsigned)f32_to_bf16_bits(vx) | ((unsigned)f32_to_bf16_bits(vy) << 16);
      pk.y = (unsigned)f32_to_bf16_bits(vz) | ((unsigned)f32_to_bf16_bits(vw) << 16);
      *(uint2*)(&As[buf][r][kc]) = pk;
    }
  };

  const int nk = Kpad / 32;
  loadA(0, 0);
  __syncthreads();

  for (int ks = 0; ks < nk; ++ks) {
    const int buf = ks & 1;
    if (ks + 1 < nk) loadA((ks + 1) * 32, buf ^ 1);   // overlap next tile with compute

    const int k0 = ks * 32;
    Frag16 a0, a1;
    a0.h[0] = *(const bf16x8*)(&As[buf][wave * 32 + l15][ka0]);
    a0.h[1] = *(const bf16x8*)(&As[buf][wave * 32 + l15][ka1]);
    a1.h[0] = *(const bf16x8*)(&As[buf][wave * 32 + 16 + l15][ka0]);
    a1.h[1] = *(const bf16x8*)(&As[buf][wave * 32 + 16 + l15][ka1]);

    #pragma unroll
    for (int t = 0; t < NT; ++t) {
      const unsigned short* bp = BT + (size_t)(t * 16 + l15) * Kpad + k0;
      Frag16 bf_;
      bf_.h[0] = *(const bf16x8*)(bp + ka0);
      bf_.h[1] = *(const bf16x8*)(bp + ka1);
      acc0[t] = __builtin_amdgcn_wmma_f32_16x16x32_bf16(
          false, a0.v, false, bf_.v, (short)0, acc0[t], false, false);
      acc1[t] = __builtin_amdgcn_wmma_f32_16x16x32_bf16(
          false, a1.v, false, bf_.v, (short)0, acc1[t], false, false);
    }
    __syncthreads();
  }

  // C layout: VGPR i -> M = mt*16 + half*8 + i, N = t*16 + l15
  for (int t = 0; t < NT; ++t) {
    int col = t * 16 + l15;
    if (col >= Nout) continue;
    #pragma unroll
    for (int i = 0; i < 8; ++i) {
      int row0 = rowBase + wave * 32 + half * 8 + i;
      int row1 = row0 + 16;
      if (row0 < M) C[(size_t)row0 * Nout + col] = acc0[t][i];
      if (row1 < M) C[(size_t)row1 * Nout + col] = acc1[t][i];
    }
  }
}

// ---------------- proj[v] = dot(X[v,:], rv)  (wave per row) ----------------
__global__ __launch_bounds__(256)
void proj_kernel(const float* __restrict__ X, const float* __restrict__ rv,
                 float* __restrict__ proj, int Nrows, int C) {
  int row  = (int)((blockIdx.x * blockDim.x + threadIdx.x) >> 5);
  int lane = threadIdx.x & 31;
  if (row >= Nrows) return;
  float s = 0.0f;
  for (int c = lane; c < C; c += 32) s += X[(size_t)row * C + c] * rv[c];
  for (int off = 16; off; off >>= 1) s += __shfl_down(s, off, 32);
  if (lane == 0) proj[row] = s;
}

// ---------------- per-edge argmax/argmin + hash-set dedup ----------------
__global__ void edge_select_kernel(const int* __restrict__ he, const float* __restrict__ proj,
                                   int* __restrict__ ea, int* __restrict__ eb,
                                   float* __restrict__ ew,
                                   unsigned long long* __restrict__ htab, int E_) {
  int e = blockIdx.x * blockDim.x + threadIdx.x;
  if (e >= E_) return;
  int v0 = he[e * 4 + 0], v1 = he[e * 4 + 1], v2 = he[e * 4 + 2], v3 = he[e * 4 + 3];
  float p0 = proj[v0], p1 = proj[v1], p2 = proj[v2], p3 = proj[v3];
  // first-max / first-min (jnp.argmax/argmin tie-break)
  int a = v0; float pa = p0;
  if (p1 > pa) { pa = p1; a = v1; }
  if (p2 > pa) { pa = p2; a = v2; }
  if (p3 > pa) { pa = p3; a = v3; }
  int b = v0; float pb = p0;
  if (p1 < pb) { pb = p1; b = v1; }
  if (p2 < pb) { pb = p2; b = v2; }
  if (p3 < pb) { pb = p3; b = v3; }
  ea[e] = a; eb[e] = b;
  float w = 0.0f;
  if (a != b) {
    int lo = a < b ? a : b;
    int hi = a < b ? b : a;
    unsigned long long key = (unsigned long long)lo * (unsigned long long)NV +
                             (unsigned long long)hi + 1ull;
    unsigned int slot = (unsigned int)((key * 0x9E3779B97F4A7C15ull) >> 43) & HMASK;
    for (;;) {
      unsigned long long old = atomicCAS(&htab[slot], 0ull, key);
      if (old == 0ull)  { w = 1.0f; break; }   // first insert -> unique winner
      if (old == key)   { w = 0.0f; break; }   // duplicate pair
      slot = (slot + 1u) & HMASK;
    }
  }
  ew[e] = w;
}

// ---------------- degree accumulation ----------------
__global__ void deg_accum_kernel(const int* __restrict__ ea, const int* __restrict__ eb,
                                 const float* __restrict__ ew, float* __restrict__ deg, int E_) {
  int e = blockIdx.x * blockDim.x + threadIdx.x;
  if (e >= E_) return;
  float w = ew[e];
  if (w != 0.0f) {
    atomicAdd(&deg[ea[e]], w);
    atomicAdd(&deg[eb[e]], w);
  }
}

__global__ void dinv_kernel(const float* __restrict__ deg, float* __restrict__ dinv, int Nv) {
  int v = blockIdx.x * blockDim.x + threadIdx.x;
  if (v < Nv) dinv[v] = rsqrtf(deg[v] + 1.0f);
}

// ---------------- feature aggregation: block per edge, thread per channel ----------------
__global__ void agg_kernel(const float* __restrict__ X, const int* __restrict__ ea,
                           const int* __restrict__ eb, const float* __restrict__ ew,
                           const float* __restrict__ dinv, float* __restrict__ agg, int C) {
  int e = blockIdx.x;
  float w = ew[e];
  if (w == 0.0f) return;
  int a = ea[e], b = eb[e];
  float coef = w * dinv[a] * dinv[b];
  int c = threadIdx.x;
  if (c < C) {
    atomicAdd(&agg[(size_t)b * C + c], X[(size_t)a * C + c] * coef);
    atomicAdd(&agg[(size_t)a * C + c], X[(size_t)b * C + c] * coef);
  }
}

// ---------------- out += X * dinv^2 (self term) ----------------
__global__ void self_add_kernel(const float* __restrict__ X, const float* __restrict__ dinv,
                                float* __restrict__ out, int Nv, int C) {
  long long i = (long long)blockIdx.x * blockDim.x + threadIdx.x;
  if (i >= (long long)Nv * C) return;
  int v = (int)(i / C);
  float d = dinv[v];
  out[i] += X[i] * d * d;
}

// ---------------- BN stats over relu(X): thread = channel, block = 512 rows ----------------
__global__ __launch_bounds__(128)
void bn_stats_kernel(const float* __restrict__ X, float* __restrict__ gsum,
                     float* __restrict__ gsq, int Nv) {
  int c  = threadIdx.x;
  int r0 = blockIdx.x * 512;
  int r1 = r0 + 512; if (r1 > Nv) r1 = Nv;
  float s = 0.0f, q = 0.0f;
  for (int r = r0; r < r1; ++r) {
    float v = X[(size_t)r * HID + c];
    v = fmaxf(v, 0.0f);
    s += v; q += v * v;
  }
  atomicAdd(&gsum[c], s);
  atomicAdd(&gsq[c], q);
}

__global__ void bn_apply_kernel(float* __restrict__ X, const float* __restrict__ gsum,
                                const float* __restrict__ gsq, const float* __restrict__ gamma,
                                const float* __restrict__ beta, int Nv) {
  long long i = (long long)blockIdx.x * blockDim.x + threadIdx.x;
  if (i >= (long long)Nv * HID) return;
  int c = (int)(i & (HID - 1));
  float inv_n = 1.0f / (float)Nv;
  float mean = gsum[c] * inv_n;
  float var  = gsq[c] * inv_n - mean * mean;
  float v = fmaxf(X[i], 0.0f);
  X[i] = (v - mean) * rsqrtf(var + 1e-5f) * gamma[c] + beta[c];
}

// ---------------- row-wise log_softmax over 40 classes (wave per row) ----------------
__global__ __launch_bounds__(256)
void log_softmax40_kernel(float* __restrict__ X, int Nv) {
  int row  = (int)((blockIdx.x * blockDim.x + threadIdx.x) >> 5);
  int lane = threadIdx.x & 31;
  if (row >= Nv) return;
  float v0 = X[(size_t)row * NCLS + lane];                 // lane 0..31 < 40 always
  bool  h1 = (lane + 32) < NCLS;
  float v1 = h1 ? X[(size_t)row * NCLS + lane + 32] : -3.402823466e38f;
  float m = fmaxf(v0, v1);
  for (int off = 16; off; off >>= 1) m = fmaxf(m, __shfl_down(m, off, 32));
  m = __shfl(m, 0, 32);
  float s = __expf(v0 - m) + (h1 ? __expf(v1 - m) : 0.0f);
  for (int off = 16; off; off >>= 1) s += __shfl_down(s, off, 32);
  s = __shfl(s, 0, 32);
  float lse = m + __logf(s);
  X[(size_t)row * NCLS + lane] = v0 - lse;
  if (h1) X[(size_t)row * NCLS + lane + 32] = v1 - lse;
}

// ---------------- launcher ----------------
extern "C" void kernel_launch(void* const* d_in, const int* in_sizes, int n_in,
                              void* d_out, int out_size, void* d_ws, size_t ws_size,
                              hipStream_t stream) {
  const float* x     = (const float*)d_in[0];
  const float* rv1   = (const float*)d_in[1];
  const float* rv2   = (const float*)d_in[2];
  const float* W1    = (const float*)d_in[3];
  const float* W2    = (const float*)d_in[4];
  const float* gamma = (const float*)d_in[5];
  const float* beta  = (const float*)d_in[6];
  const int*   he    = (const int*)d_in[7];
  float* out = (float*)d_out;

  // workspace carve (256B aligned chunks)
  char* p = (char*)d_ws;
  auto carve = [&](size_t bytes) -> void* {
    void* r = (void*)p;
    p += (bytes + 255) & ~(size_t)255;
    return r;
  };
  float* h    = (float*)carve((size_t)NV * HID * 4);
  float* s1   = (float*)carve((size_t)NV * HID * 4);
  float* o    = (float*)carve((size_t)NV * NCLS * 4);
  float* proj = (float*)carve((size_t)NV * 4);
  int*   ea   = (int*)  carve((size_t)NE * 4);
  int*   eb   = (int*)  carve((size_t)NE * 4);
  float* ew   = (float*)carve((size_t)NE * 4);
  float* deg  = (float*)carve((size_t)NV * 4);
  float* dinv = (float*)carve((size_t)NV * 4);
  unsigned long long* htab = (unsigned long long*)carve((size_t)HSIZE * 8);
  float* gsum = (float*)carve(512);
  float* gsq  = (float*)carve(512);
  unsigned short* W1bT = (unsigned short*)carve((size_t)HID * 512 * 2);  // [128 n][512 k]
  unsigned short* W2bT = (unsigned short*)carve((size_t)48 * HID * 2);   // [48 n][128 k]

  const int mBlocks = (NV + 255) / 256;          // 782 (256-row block tiles)
  const int eBlocks = (NE + 255) / 256;
  const int vBlocks = (NV + 255) / 256;
  const int waveRowBlocks = (NV + 7) / 8;        // wave-per-row kernels, 8 waves/block

  // -------- weight preconversion (transposed, padded) --------
  cvt_wT_kernel<<<(HID * 512 + 255) / 256, 256, 0, stream>>>(W1, W1bT, INCH, 512, HID, HID);
  cvt_wT_kernel<<<(48 * HID + 255) / 256, 256, 0, stream>>>(W2, W2bT, HID, HID, NCLS, 48);

  // -------- layer 1: h = x @ W1 --------
  gemm_bf16_wmma<8><<<mBlocks, 256, 0, stream>>>(x, W1bT, h, NV, INCH, 512, HID);

  // -------- smooth(h, rv1) -> s1 --------
  proj_kernel<<<waveRowBlocks, 256, 0, stream>>>(h, rv1, proj, NV, HID);
  hipMemsetAsync(htab, 0, (size_t)HSIZE * 8, stream);
  hipMemsetAsync(deg, 0, (size_t)NV * 4, stream);
  edge_select_kernel<<<eBlocks, 256, 0, stream>>>(he, proj, ea, eb, ew, htab, NE);
  deg_accum_kernel<<<eBlocks, 256, 0, stream>>>(ea, eb, ew, deg, NE);
  dinv_kernel<<<vBlocks, 256, 0, stream>>>(deg, dinv, NV);
  hipMemsetAsync(s1, 0, (size_t)NV * HID * 4, stream);
  agg_kernel<<<NE, HID, 0, stream>>>(h, ea, eb, ew, dinv, s1, HID);
  self_add_kernel<<<(int)(((long long)NV * HID + 255) / 256), 256, 0, stream>>>(h, dinv, s1, NV, HID);

  // -------- relu + batchnorm (in place on s1) --------
  hipMemsetAsync(gsum, 0, 512, stream);
  hipMemsetAsync(gsq, 0, 512, stream);
  bn_stats_kernel<<<(NV + 511) / 512, HID, 0, stream>>>(s1, gsum, gsq, NV);
  bn_apply_kernel<<<(int)(((long long)NV * HID + 255) / 256), 256, 0, stream>>>(s1, gsum, gsq, gamma, beta, NV);

  // -------- layer 2: o = s1 @ W2 --------
  gemm_bf16_wmma<3><<<mBlocks, 256, 0, stream>>>(s1, W2bT, o, NV, HID, HID, NCLS);

  // -------- smooth(o, rv2) -> out --------
  proj_kernel<<<waveRowBlocks, 256, 0, stream>>>(o, rv2, proj, NV, NCLS);
  hipMemsetAsync(htab, 0, (size_t)HSIZE * 8, stream);
  hipMemsetAsync(deg, 0, (size_t)NV * 4, stream);
  edge_select_kernel<<<eBlocks, 256, 0, stream>>>(he, proj, ea, eb, ew, htab, NE);
  deg_accum_kernel<<<eBlocks, 256, 0, stream>>>(ea, eb, ew, deg, NE);
  dinv_kernel<<<vBlocks, 256, 0, stream>>>(deg, dinv, NV);
  hipMemsetAsync(out, 0, (size_t)NV * NCLS * 4, stream);
  agg_kernel<<<NE, 64, 0, stream>>>(o, ea, eb, ew, dinv, out, NCLS);
  self_add_kernel<<<(int)(((long long)NV * NCLS + 255) / 256), 256, 0, stream>>>(o, dinv, out, NV, NCLS);

  // -------- log_softmax (in place on out) --------
  log_softmax40_kernel<<<waveRowBlocks, 256, 0, stream>>>(out, NV);
}
